// BlindCrossAttention_36146444763871
// MI455X (gfx1250) — compile-verified, matrix operand order checked by reading, not run on hardware
//
#include <hip/hip_runtime.h>
#include <math.h>

// ---------------------------------------------------------------------------
// BlindCrossAttention for MI455X (gfx1250, wave32, WMMA 16x16x32 f16->f32)
//
// Roofline: ~125 GFLOP of GEMM vs ~300MB of traffic (att output alone is
// 134MB and is mandatory) -> compute bound -> run every matmul on
// v_wmma_f32_16x16x32_f16. Softmax/LN stay in f32 in LDS (128KB score
// panel per 16-row block fits easily in the 320KB WGP LDS).
// Inner K-loops are kept as single basic blocks so the scheduler software-
// pipelines clause'd global_load_b128 pairs across WMMAs; panel warm-up
// prefetches (global_prefetch_b8) are hoisted in front of the loops.
// ---------------------------------------------------------------------------

typedef __attribute__((ext_vector_type(16))) _Float16 v16h;
typedef __attribute__((ext_vector_type(8)))  float    v8f;

#define NB   4
#define NQ   2048
#define NK   2048
#define DD   1024
#define PP   64

// ---- WMMA fragment loaders -------------------------------------------------
// ISA 7.12.2, 16-bit A-matrix 16x32: lanes 0-15 hold M=lane with halves
// 0..7 = K 0..7 and halves 8..15 = K 16..23; lanes 16-31 hold M=lane-16 with
// K 8..15 / 24..31.  So each lane does two contiguous 16-byte loads at
// k-offset base0 and base0+16 where base0 = (lane>=16) ? 8 : 0.
// B-operand uses the same per-lane K packing with lane selecting column N,
// i.e. lane n loads row n of the transposed matrix.
static __device__ __forceinline__ v16h frag_f16(const _Float16* __restrict__ src,
                                                int ld, int row0, int k0, int lane) {
  const _Float16* p = src + (size_t)(row0 + (lane & 15)) * ld + (k0 + ((lane >> 4) << 3));
  union { v16h v; int4 q[2]; } u;
  u.q[0] = *(const int4*)(p);
  u.q[1] = *(const int4*)(p + 16);
  return u.v;
}

// Same layout, but source is f32 (global or LDS) converted on the fly.
static __device__ __forceinline__ v16h frag_f32cvt(const float* __restrict__ src,
                                                   int ld, int row0, int k0, int lane) {
  const float* p = src + (size_t)(row0 + (lane & 15)) * ld + (k0 + ((lane >> 4) << 3));
  v16h v;
#pragma unroll
  for (int i = 0; i < 8; ++i) {
    v[i]     = (_Float16)p[i];
    v[i + 8] = (_Float16)p[i + 16];
  }
  return v;
}

// One-shot panel warm-up prefetch (global_prefetch_b8); issued OUTSIDE the
// K-loops so it never breaks the single-basic-block pipelined schedule.
static __device__ __forceinline__ void prefetch_frag(const _Float16* __restrict__ src,
                                                     int ld, int row0, int k0, int lane) {
  const _Float16* p = src + (size_t)(row0 + (lane & 15)) * ld + (k0 + ((lane >> 4) << 3));
  __builtin_prefetch((const void*)p, 0, 1);
}

static __device__ __forceinline__ v8f wmma16(v16h a, v16h b, v8f c) {
  // D = A(16x32) * B(32x16) + C, f32 accumulate
  return __builtin_amdgcn_wmma_f32_16x16x32_f16(false, a, false, b, (short)0, c,
                                                false, false);
}

static __device__ __forceinline__ float wave_sum(float v) {
#pragma unroll
  for (int m = 16; m >= 1; m >>= 1) v += __shfl_xor(v, m, 32);
  return v;
}
static __device__ __forceinline__ float wave_max(float v) {
#pragma unroll
  for (int m = 16; m >= 1; m >>= 1) v = fmaxf(v, __shfl_xor(v, m, 32));
  return v;
}

// ---------------------------------------------------------------------------
// K1: f32 [R][C] -> f16 transposed [C][R]  (WMMA B operands)
// ---------------------------------------------------------------------------
__global__ void cvt_transpose_kernel(const float* __restrict__ src,
                                     _Float16* __restrict__ dst, int R, int C) {
  int idx = blockIdx.x * blockDim.x + threadIdx.x;
  int r = idx / C, c = idx - r * C;
  dst[(size_t)c * R + r] = (_Float16)src[idx];
}

// ---------------------------------------------------------------------------
// K2: sinusoidal position table: pos[j][i]=sin(j*10000^-(i/32)) (i<32),
//     pos[j][32+i]=cos(...).  Writes pos (row major) and posT.
// ---------------------------------------------------------------------------
__global__ void pos_kernel(_Float16* __restrict__ pos, _Float16* __restrict__ posT) {
  int j = blockIdx.x * blockDim.x + threadIdx.x;  // 0..2047
  float pj = (float)j;
#pragma unroll
  for (int i = 0; i < 32; ++i) {
    float e  = (float)i * (1.0f / 32.0f);
    float f  = __expf(-e * 9.210340371976184f);   // 10000^-e
    float ph = pj * f;
    float s  = __sinf(ph);
    float c  = __cosf(ph);
    pos[(size_t)j * PP + i]        = (_Float16)s;
    pos[(size_t)j * PP + 32 + i]   = (_Float16)c;
    posT[(size_t)i * NK + j]       = (_Float16)s;
    posT[(size_t)(i + 32) * NK + j] = (_Float16)c;
  }
}

// ---------------------------------------------------------------------------
// K3: Y = LayerNorm(X @ W + bias)   (W given transposed, f16)
//     One WG = 16 rows; wave w computes cols [w*128, w*128+128) via WMMA,
//     result staged in LDS (16x1024 f32), then full-row LN.
//     Y (row-major f16) and/or YT ([b][d][2048] f16) outputs.
// Dynamic LDS: 16*1024*4 = 65536 bytes.
// ---------------------------------------------------------------------------
__global__ void proj_ln_kernel(const float* __restrict__ X,
                               const _Float16* __restrict__ Wt,
                               const float* __restrict__ bias,
                               const float* __restrict__ gamma,
                               const float* __restrict__ beta,
                               _Float16* __restrict__ Y,
                               _Float16* __restrict__ YT) {
  extern __shared__ char smem[];
  float* S = (float*)smem;  // [16][1024]
  const int lane = threadIdx.x & 31, wave = threadIdx.x >> 5;
  const int row0 = blockIdx.x << 4;
  const int c0 = wave << 7;

  // warm the B-panel stream (one-shot, outside the pipelined loop)
  prefetch_frag(Wt, DD, c0, 256, lane);
  prefetch_frag(Wt, DD, c0 + 64, 256, lane);
  prefetch_frag(Wt, DD, c0, 512, lane);
  prefetch_frag(Wt, DD, c0 + 64, 512, lane);

  v8f zero = {};
  v8f acc[8];
#pragma unroll
  for (int t = 0; t < 8; ++t) acc[t] = zero;

  for (int kb = 0; kb < DD; kb += 32) {
    v16h a = frag_f32cvt(X, DD, row0, kb, lane);
#pragma unroll
    for (int t = 0; t < 8; ++t) {
      v16h bf = frag_f16(Wt, DD, c0 + (t << 4), kb, lane);
      acc[t] = wmma16(a, bf, acc[t]);
    }
  }

  const int rbase = (lane >> 4) << 3;  // C/D layout: lanes16-31 hold M=8..15
#pragma unroll
  for (int t = 0; t < 8; ++t) {
    int col = c0 + (t << 4) + (lane & 15);
    float bv = bias[col];
#pragma unroll
    for (int r = 0; r < 8; ++r) S[(rbase + r) * DD + col] = acc[t][r] + bv;
  }
  __syncthreads();

  // LayerNorm: wave w owns rows 2w, 2w+1
  for (int rr = wave * 2; rr < wave * 2 + 2; ++rr) {
    float s = 0.f, s2 = 0.f;
    for (int c = lane; c < DD; c += 32) {
      float v = S[rr * DD + c];
      s += v; s2 += v * v;
    }
    s = wave_sum(s); s2 = wave_sum(s2);
    float mean = s * (1.0f / DD);
    float var  = s2 * (1.0f / DD) - mean * mean;
    float rstd = rsqrtf(var + 1e-5f);
    size_t m = (size_t)row0 + rr;
    for (int c = lane; c < DD; c += 32) {
      float v = (S[rr * DD + c] - mean) * rstd * gamma[c] + beta[c];
      if (Y)  Y[m * DD + c] = (_Float16)v;
      if (YT) {
        int bb = (int)(m >> 11), n = (int)(m & 2047);
        YT[(((size_t)bb << 10) + c) * (size_t)NK + n] = (_Float16)v;
      }
    }
  }
}

// ---------------------------------------------------------------------------
// K4: attention 1 + pos_net.
//   S1 = qn @ kn^T / 32 -> softmax -> a1 (written to att[h=0]) -> x = a1@pos
//   -> x' = x@Wp + bp  (f16 into ws).
// One WG = (batch, 16 query rows). Scores in LDS 16x2048 f32.
// Dynamic LDS: 131072 + 32768 + 4096 = 167936 bytes.
// ---------------------------------------------------------------------------
__global__ void att1_kernel(const _Float16* __restrict__ qn,
                            const _Float16* __restrict__ kn,
                            const _Float16* __restrict__ posT,
                            const _Float16* __restrict__ WpT,
                            const float* __restrict__ bp,
                            float* __restrict__ att,
                            _Float16* __restrict__ xh) {
  extern __shared__ char smem[];
  float* S  = (float*)smem;                              // [16][2048]
  float* XP = (float*)(smem + 16 * NK * 4);              // [8][16][64]
  float* XS = (float*)(smem + 16 * NK * 4 + 8 * 16 * 64 * 4);  // [16][64]
  const int lane = threadIdx.x & 31, wave = threadIdx.x >> 5;
  const int b  = blockIdx.x >> 7;
  const int r0 = (blockIdx.x & 127) << 4;
  const int qrow = b * NQ + r0;
  const int c0 = wave << 8;          // 256 score columns per wave
  const int rbase = (lane >> 4) << 3;

  // warm the kn panel stream
  prefetch_frag(kn, DD, b * NK + c0, 256, lane);
  prefetch_frag(kn, DD, b * NK + c0 + 128, 256, lane);
  prefetch_frag(kn, DD, b * NK + c0, 512, lane);
  prefetch_frag(kn, DD, b * NK + c0 + 128, 512, lane);

  v8f zero = {};
  v8f acc[16];
#pragma unroll
  for (int t = 0; t < 16; ++t) acc[t] = zero;

  for (int kb = 0; kb < DD; kb += 32) {
    v16h a = frag_f16(qn, DD, qrow, kb, lane);
#pragma unroll
    for (int t = 0; t < 16; ++t) {
      v16h bf = frag_f16(kn, DD, b * NK + c0 + (t << 4), kb, lane);
      acc[t] = wmma16(a, bf, acc[t]);
    }
  }
#pragma unroll
  for (int t = 0; t < 16; ++t) {
    int col = c0 + (t << 4) + (lane & 15);
#pragma unroll
    for (int r = 0; r < 8; ++r) S[(rbase + r) * NK + col] = acc[t][r] * 0.03125f;
  }
  __syncthreads();

  // exact softmax + write a1 (h=0)
  for (int rr = wave * 2; rr < wave * 2 + 2; ++rr) {
    float mx = -3.4e38f;
    for (int c = lane; c < NK; c += 32) mx = fmaxf(mx, S[rr * NK + c]);
    mx = wave_max(mx);
    float sum = 0.f;
    for (int c = lane; c < NK; c += 32) {
      float e = __expf(S[rr * NK + c] - mx);
      S[rr * NK + c] = e;
      sum += e;
    }
    sum = wave_sum(sum);
    float inv = 1.0f / sum;
    float* arow = att + (((size_t)b * 2) * NQ + (r0 + rr)) * (size_t)NK;
    for (int c = lane; c < NK; c += 32) {
      float p = S[rr * NK + c] * inv;
      S[rr * NK + c] = p;
      arow[c] = p;
    }
  }
  __syncthreads();

  // x = P @ pos : split-K across waves (wave w handles K in [c0, c0+256))
  v8f ax[4];
#pragma unroll
  for (int t = 0; t < 4; ++t) ax[t] = zero;
  for (int kb = c0; kb < c0 + 256; kb += 32) {
    v16h a = frag_f32cvt(S, NK, 0, kb, lane);
#pragma unroll
    for (int t = 0; t < 4; ++t) {
      v16h bf = frag_f16(posT, NK, t << 4, kb, lane);
      ax[t] = wmma16(a, bf, ax[t]);
    }
  }
#pragma unroll
  for (int t = 0; t < 4; ++t) {
    int col = (t << 4) + (lane & 15);
#pragma unroll
    for (int r = 0; r < 8; ++r) XP[wave * 1024 + (rbase + r) * PP + col] = ax[t][r];
  }
  __syncthreads();
  for (int idx = threadIdx.x; idx < 16 * PP; idx += 256) {
    float s = 0.f;
#pragma unroll
    for (int w = 0; w < 8; ++w) s += XP[w * 1024 + idx];
    XS[idx] = s;
  }
  __syncthreads();

  // pos_net: x' = XS(16x64) @ Wp + bp ; waves 0..3, one 16x16 tile each
  if (wave < 4) {
    v8f ap = zero;
#pragma unroll
    for (int kb = 0; kb < PP; kb += 32) {
      v16h a  = frag_f32cvt(XS, PP, 0, kb, lane);
      v16h bf = frag_f16(WpT, PP, wave << 4, kb, lane);
      ap = wmma16(a, bf, ap);
    }
    int col = (wave << 4) + (lane & 15);
    float bpv = bp[col];
#pragma unroll
    for (int r = 0; r < 8; ++r) {
      size_t m = (size_t)qrow + rbase + r;
      xh[m * PP + col] = (_Float16)(ap[r] + bpv);
    }
  }
}

// ---------------------------------------------------------------------------
// K5: attention 2.
//   S2 = x' @ pos^T / 8 -> softmax -> a2 (att[h=1]) -> out = a2 @ vn (via vnT)
// Dynamic LDS: 131072 bytes.
// ---------------------------------------------------------------------------
__global__ void att2_kernel(const _Float16* __restrict__ xh,
                            const _Float16* __restrict__ pos,
                            const _Float16* __restrict__ vnT,
                            float* __restrict__ att,
                            float* __restrict__ out) {
  extern __shared__ char smem[];
  float* S = (float*)smem;  // [16][2048]
  const int lane = threadIdx.x & 31, wave = threadIdx.x >> 5;
  const int b  = blockIdx.x >> 7;
  const int r0 = (blockIdx.x & 127) << 4;
  const int qrow = b * NQ + r0;
  const int c0 = wave << 8;
  const int rbase = (lane >> 4) << 3;

  v8f zero = {};
  v8f acc[16];
#pragma unroll
  for (int t = 0; t < 16; ++t) acc[t] = zero;

#pragma unroll
  for (int kb = 0; kb < PP; kb += 32) {
    v16h a = frag_f16(xh, PP, qrow, kb, lane);
#pragma unroll
    for (int t = 0; t < 16; ++t) {
      v16h bf = frag_f16(pos, PP, c0 + (t << 4), kb, lane);
      acc[t] = wmma16(a, bf, acc[t]);
    }
  }
#pragma unroll
  for (int t = 0; t < 16; ++t) {
    int col = c0 + (t << 4) + (lane & 15);
#pragma unroll
    for (int r = 0; r < 8; ++r) S[(rbase + r) * NK + col] = acc[t][r] * 0.125f;
  }
  __syncthreads();

  for (int rr = wave * 2; rr < wave * 2 + 2; ++rr) {
    float mx = -3.4e38f;
    for (int c = lane; c < NK; c += 32) mx = fmaxf(mx, S[rr * NK + c]);
    mx = wave_max(mx);
    float sum = 0.f;
    for (int c = lane; c < NK; c += 32) {
      float e = __expf(S[rr * NK + c] - mx);
      S[rr * NK + c] = e;
      sum += e;
    }
    sum = wave_sum(sum);
    float inv = 1.0f / sum;
    float* arow = att + (((size_t)b * 2 + 1) * NQ + (r0 + rr)) * (size_t)NK;
    for (int c = lane; c < NK; c += 32) {
      float p = S[rr * NK + c] * inv;
      S[rr * NK + c] = p;
      arow[c] = p;
    }
  }
  __syncthreads();

  // out = P(16x2048) @ vn(2048x1024): wave w -> output cols [w*128, w*128+128)
  const _Float16* vb = vnT + ((size_t)b << 10) * (size_t)NK;

  // warm the vn panel stream
  prefetch_frag(vb, NK, (wave << 7), 256, lane);
  prefetch_frag(vb, NK, (wave << 7) + 64, 256, lane);
  prefetch_frag(vb, NK, (wave << 7), 512, lane);
  prefetch_frag(vb, NK, (wave << 7) + 64, 512, lane);

  v8f ao[8];
#pragma unroll
  for (int t = 0; t < 8; ++t) ao[t] = zero;
  for (int kb = 0; kb < NK; kb += 32) {
    v16h a = frag_f32cvt(S, NK, 0, kb, lane);
#pragma unroll
    for (int t = 0; t < 8; ++t) {
      v16h bf = frag_f16(vb, NK, (wave << 7) + (t << 4), kb, lane);
      ao[t] = wmma16(a, bf, ao[t]);
    }
  }
#pragma unroll
  for (int t = 0; t < 8; ++t) {
    int col = (wave << 7) + (t << 4) + (lane & 15);
#pragma unroll
    for (int r = 0; r < 8; ++r)
      out[((size_t)qrow + rbase + r) * DD + col] = ao[t][r];
  }
}

// ---------------------------------------------------------------------------
extern "C" void kernel_launch(void* const* d_in, const int* in_sizes, int n_in,
                              void* d_out, int out_size, void* d_ws, size_t ws_size,
                              hipStream_t stream) {
  (void)in_sizes; (void)n_in; (void)out_size; (void)ws_size;
  const float* q     = (const float*)d_in[0];
  const float* k     = (const float*)d_in[1];
  const float* Wq    = (const float*)d_in[2];
  const float* bq    = (const float*)d_in[3];
  const float* Wk    = (const float*)d_in[4];
  const float* bk    = (const float*)d_in[5];
  const float* Wv    = (const float*)d_in[6];
  const float* bv    = (const float*)d_in[7];
  const float* gq    = (const float*)d_in[8];
  const float* betaq = (const float*)d_in[9];
  const float* gk    = (const float*)d_in[10];
  const float* betak = (const float*)d_in[11];
  const float* gv    = (const float*)d_in[12];
  const float* betav = (const float*)d_in[13];
  const float* Wp    = (const float*)d_in[14];
  const float* bp    = (const float*)d_in[15];

  // workspace carve (f16 staging buffers, 256B aligned)
  char* ws = (char*)d_ws;
  size_t off = 0;
  auto take = [&](size_t bytes) -> char* {
    char* p = ws + off;
    off = (off + bytes + 255) & ~(size_t)255;
    return p;
  };
  _Float16* Wqt  = (_Float16*)take((size_t)DD * DD * 2);
  _Float16* Wkt  = (_Float16*)take((size_t)DD * DD * 2);
  _Float16* Wvt  = (_Float16*)take((size_t)DD * DD * 2);
  _Float16* WpT  = (_Float16*)take((size_t)PP * PP * 2);
  _Float16* posH = (_Float16*)take((size_t)NK * PP * 2);
  _Float16* posT = (_Float16*)take((size_t)PP * NK * 2);
  _Float16* qnH  = (_Float16*)take((size_t)NB * NQ * DD * 2);
  _Float16* knH  = (_Float16*)take((size_t)NB * NK * DD * 2);
  _Float16* vnT  = (_Float16*)take((size_t)NB * DD * NK * 2);
  _Float16* xH   = (_Float16*)take((size_t)NB * NQ * PP * 2);

  float* outp = (float*)d_out;                              // [B,NQ,D]
  float* attp = outp + (size_t)NB * NQ * DD;                // [B,2,NQ,NK]

  // 1) weight transposes to f16
  cvt_transpose_kernel<<<(DD * DD) / 256, 256, 0, stream>>>(Wq, Wqt, DD, DD);
  cvt_transpose_kernel<<<(DD * DD) / 256, 256, 0, stream>>>(Wk, Wkt, DD, DD);
  cvt_transpose_kernel<<<(DD * DD) / 256, 256, 0, stream>>>(Wv, Wvt, DD, DD);
  cvt_transpose_kernel<<<(PP * PP) / 256, 256, 0, stream>>>(Wp, WpT, PP, PP);

  // 2) sinusoidal positions
  pos_kernel<<<NK / 256, 256, 0, stream>>>(posH, posT);

  // 3) projections + LayerNorm (16 rows per WG, 64KB dynamic LDS)
  const int projGrid = (NB * NQ) / 16;
  proj_ln_kernel<<<projGrid, 256, 65536, stream>>>(q, Wqt, bq, gq, betaq, qnH, nullptr);
  proj_ln_kernel<<<projGrid, 256, 65536, stream>>>(k, Wkt, bk, gk, betak, knH, nullptr);
  proj_ln_kernel<<<projGrid, 256, 65536, stream>>>(k, Wvt, bv, gv, betav, nullptr, vnT);

  // 4) attention 1 + pos_net  (167936B dynamic LDS)
  att1_kernel<<<NB * (NQ / 16), 256, 167936, stream>>>(qnH, knH, posT, WpT, bp, attp, xH);

  // 5) attention 2 + output GEMM (131072B dynamic LDS)
  att2_kernel<<<NB * (NQ / 16), 256, 131072, stream>>>(xH, posH, vnT, attp, outp);
}
